// DeformConvV3_12214886990431
// MI455X (gfx1250) — matrix operand order repaired and checked by reading, not compiled, not run on hardware
//
#include <hip/hip_runtime.h>

// ---------------------------------------------------------------------------
// Types for CDNA5 WMMA (wave32, gfx1250)
// ---------------------------------------------------------------------------
typedef __attribute__((ext_vector_type(16))) __bf16 v16bf;
typedef __attribute__((ext_vector_type(8)))  float  v8f;
typedef __attribute__((ext_vector_type(4)))  unsigned int u32x4;
typedef __attribute__((ext_vector_type(8)))  unsigned int u32x8;

union FragBF16 {            // 32-byte WMMA operand fragment (16 x bf16)
    uint4 q[2];
    v16bf v;
};

__device__ __forceinline__ unsigned short f2bf(float f) {
    unsigned int u = __float_as_uint(f);
    u += 0x7FFFu + ((u >> 16) & 1u);    // round-to-nearest-even
    return (unsigned short)(u >> 16);
}

// Problem constants
#define CC   256
#define GG   8
#define GCH  32
#define HH   56
#define WW   56
#define BB   8
#define HWSZ (HH * WW)            // 3136
#define MTOT (BB * HWSZ)          // 25088
#define NOM  224                  // 144 offset cols + 72 mask cols + 8 pad
#define KPAD 264                  // 256 + 8 bf16 pad (16B) -> conflict-free LDS

// ---------------------------------------------------------------------------
// Tensor Data Mover: 1-D contiguous global -> LDS copy with hardware padding.
// n8 = size in 8-byte units. TDM inserts 16B of LDS padding after every 512B
// (pad_interval code 6 = 128 DWORDs, pad_amount code 3 = 4 DWORDs), which
// reproduces the KPAD=264-element row layout used by the WMMA fragment reads.
// D# layout per CDNA5 ISA (groups 0 and 1 only;
// VADDR2/VADDR3 omitted => tensor up to 2D). Tracked by TENSORcnt.
// ---------------------------------------------------------------------------
__device__ __forceinline__ void tdm_load_1d_padded(unsigned lds_addr,
                                                   const void* gptr,
                                                   unsigned n8) {
    unsigned long long ga = (unsigned long long)(size_t)gptr;
    u32x4 g0;
    g0[0] = 1u;                                    // count=1, user-mode load
    g0[1] = lds_addr;                              // LDS byte address
    g0[2] = (unsigned)ga;                          // global_addr[31:0]
    g0[3] = (unsigned)((ga >> 32) & 0x01FFFFFFu)   // global_addr[56:32]
          | (2u << 30);                            // type = 2 ("image")
    u32x8 g1;
    g1[0] = (3u << 16)    // data_size = 8 bytes
          | (1u << 20)    // pad_enable
          | (6u << 22)    // pad_interval code 6 -> every 128 DWORDs (512B)
          | (3u << 25);   // pad_amount  code 3 -> 4 DWORDs (16B)
    g1[1] = (n8 & 0xFFFFu) << 16;                  // tensor_dim0[15:0]
    g1[2] = (n8 >> 16) | (1u << 16);               // tensor_dim0[31:16] | tensor_dim1=1
    g1[3] = (n8 << 16);                            // tensor_dim1 hi=0 | tile_dim0
    g1[4] = 1u;                                    // tile_dim1 = 1, tile_dim2 = 0
    g1[5] = n8;                                    // tensor_dim0_stride lo
    g1[6] = 0u;                                    // stride hi | dim1_stride lo
    g1[7] = 0u;                                    // dim1_stride hi
    asm volatile("tensor_load_to_lds %0, %1" :: "s"(g0), "s"(g1) : "memory");
}

// ---------------------------------------------------------------------------
// Weight prep: wT[n*256 + k] = bf16( w[k*N + n] )
// ---------------------------------------------------------------------------
__global__ __launch_bounds__(256) void prep_wT_kernel(
    const float* __restrict__ w, unsigned short* __restrict__ wT, int N) {
    int idx = blockIdx.x * 256 + threadIdx.x;
    if (idx >= N * 256) return;
    int n = idx >> 8, k = idx & 255;
    wT[idx] = f2bf(w[(size_t)k * N + n]);
}

__global__ __launch_bounds__(256) void prep_om_tail_kernel(
    const float* __restrict__ b_off, const float* __restrict__ b_mask,
    unsigned short* __restrict__ wOMT, float* __restrict__ bias_om) {
    int i = blockIdx.x * 256 + threadIdx.x;
    // zero-fill pad rows 216..223 of wOMT (8*256 = 2048 entries)
    if (i < 8 * 256) wOMT[216 * 256 + i] = 0;
    // fused bias
    if (i < 144)            bias_om[i] = b_off[i];
    else if (i < 216)       bias_om[i] = b_mask[i - 144];
    else if (i < NOM)       bias_om[i] = 0.0f;
}

// ---------------------------------------------------------------------------
// x NCHW f32 -> NHWC bf16 via LDS tile transpose (coalesced both ways)
// ---------------------------------------------------------------------------
__global__ __launch_bounds__(256) void transpose_x_kernel(
    const float* __restrict__ x, unsigned short* __restrict__ xT) {
    __shared__ float tile[32][33];
    int b   = blockIdx.z;
    int c0  = blockIdx.y * 32;
    int hw0 = blockIdx.x * 32;
    int t = threadIdx.x, col = t & 31, rq = t >> 5;
#pragma unroll
    for (int k = 0; k < 4; ++k) {
        int r = rq + k * 8;  // channel row in tile
        tile[r][col] = x[((size_t)(b * CC + c0 + r)) * HWSZ + hw0 + col];
    }
    __syncthreads();
#pragma unroll
    for (int k = 0; k < 4; ++k) {
        int r = rq + k * 8;  // hw row
        xT[((size_t)(b * HWSZ + hw0 + r)) * CC + c0 + col] = f2bf(tile[col][r]);
    }
}

// ---------------------------------------------------------------------------
// Depthwise 3x3 conv (NCHW in), writes fp32 NHWC tmp
// grid = (H, B*C), block = 64 (2 waves), coalesced along W
// ---------------------------------------------------------------------------
__global__ __launch_bounds__(64) void dw_conv_kernel(
    const float* __restrict__ x, const float* __restrict__ dww,
    const float* __restrict__ dwb, float* __restrict__ ftmp) {
    int w = threadIdx.x;
    if (w >= WW) return;
    int h  = blockIdx.x;
    int bc = blockIdx.y;
    int b = bc >> 8, c = bc & 255;
    const float* xp = x + (size_t)bc * HWSZ;
    const float* wp = dww + c * 9;
    float s = dwb[c];
#pragma unroll
    for (int ky = 0; ky < 3; ++ky) {
        int yy = h + ky - 1;
        if ((unsigned)yy >= (unsigned)HH) continue;
#pragma unroll
        for (int kx = 0; kx < 3; ++kx) {
            int xx = w + kx - 1;
            if ((unsigned)xx >= (unsigned)WW) continue;
            s = fmaf(xp[yy * WW + xx], wp[ky * 3 + kx], s);
        }
    }
    ftmp[((size_t)(b * HWSZ) + h * WW + w) * CC + c] = s;
}

// ---------------------------------------------------------------------------
// LayerNorm(C) + exact GELU per pixel; 256 threads = 8 waves; writes bf16
// ---------------------------------------------------------------------------
__global__ __launch_bounds__(256) void ln_gelu_kernel(
    const float* __restrict__ ftmp, const float* __restrict__ lg,
    const float* __restrict__ lb, unsigned short* __restrict__ fbf) {
    int m = blockIdx.x, c = threadIdx.x;
    float v = ftmp[(size_t)m * CC + c];
    float s1 = v, s2 = v * v;
#pragma unroll
    for (int o = 16; o >= 1; o >>= 1) {
        s1 += __shfl_xor(s1, o);
        s2 += __shfl_xor(s2, o);
    }
    __shared__ float red[16];
    int wave = c >> 5, lane = c & 31;
    if (lane == 0) { red[wave] = s1; red[8 + wave] = s2; }
    __syncthreads();
    float t1 = 0.f, t2 = 0.f;
#pragma unroll
    for (int i = 0; i < 8; ++i) { t1 += red[i]; t2 += red[8 + i]; }
    float mean = t1 * (1.0f / CC);
    float var  = t2 * (1.0f / CC) - mean * mean;
    float xn = (v - mean) * rsqrtf(var + 1e-5f) * lg[c] + lb[c];
    float ge = 0.5f * xn * (1.0f + erff(xn * 0.70710678118654752f));
    fbf[(size_t)m * CC + c] = f2bf(ge);
}

// ---------------------------------------------------------------------------
// WMMA GEMM: out[M,N] = A[M,256](bf16) * WT[N,256]^T(bf16) + bias
// Block: 256 thr (8 waves as 4x2 of 16x16 WMMA tiles) -> 64x32 tile.
// A (64x256) and B (32x256) tiles are contiguous in global memory and are
// staged into padded LDS by the Tensor Data Mover (wave 0 issues two
// tensor_load_to_lds ops, waits TENSORcnt, barrier publishes to all waves).
// Then 8 x v_wmma_f32_16x16x32_bf16 per wave over K=256.
// MODE 0: plain f32 [M,N] store.  MODE 2: +BN +SiLU, NCHW scatter.
// ---------------------------------------------------------------------------
template <int MODE>
__global__ __launch_bounds__(256) void gemm_bf16_kernel(
    const unsigned short* __restrict__ A, const unsigned short* __restrict__ WT,
    const float* __restrict__ bias, float* __restrict__ out, int N,
    const float* __restrict__ bn_g, const float* __restrict__ bn_b,
    const float* __restrict__ bn_mean, const float* __restrict__ bn_var) {
    constexpr int K = 256;
    __shared__ unsigned short sA[64 * KPAD];  // 33 KB, +16B row pad via TDM
    __shared__ unsigned short sB[32 * KPAD];  // 16.5 KB

    const int t  = threadIdx.x;
    const int m0 = blockIdx.y * 64;
    const int n0 = blockIdx.x * 32;

    const int wave = t >> 5, lane = t & 31;
    if (wave == 0) {
        // Async DMA stage of both operand tiles; LDS padding done by TDM.
        tdm_load_1d_padded((unsigned)(size_t)sA, A + (size_t)m0 * K,
                           (64u * K * 2u) / 8u);
        tdm_load_1d_padded((unsigned)(size_t)sB, WT + (size_t)n0 * K,
                           (32u * K * 2u) / 8u);
        __builtin_amdgcn_s_wait_tensorcnt(0);
    }
    __syncthreads();

    const int wm = wave >> 1, wn = wave & 1;
    const int lr = lane & 15, hi = lane >> 4;

    v8f acc = {};
    const uint4* aRow = reinterpret_cast<const uint4*>(sA + (wm * 16 + lr) * KPAD);
    const uint4* bRow = reinterpret_cast<const uint4*>(sB + (wn * 16 + lr) * KPAD);
#pragma unroll
    for (int ks = 0; ks < 8; ++ks) {
        // A lane: K = {8*hi .. 8*hi+7} U {16+8*hi .. 16+8*hi+7} within 32-chunk
        FragBF16 fa;
        fa.q[0] = aRow[ks * 4 + hi];
        fa.q[1] = aRow[ks * 4 + 2 + hi];
        // B lane: K = {8*hi .. 8*hi+7} U {16+8*hi .. 16+8*hi+7} within 32-chunk
        FragBF16 fb;
        fb.q[0] = bRow[ks * 4 + hi];
        fb.q[1] = bRow[ks * 4 + 2 + hi];
        acc = __builtin_amdgcn_wmma_f32_16x16x32_bf16(
            false, fa.v, false, fb.v, (short)0, acc, false, false);
    }

    const int n  = n0 + wn * 16 + lr;
    const float bv = bias[n];
    float sc = 0.f, sh = 0.f;
    if (MODE == 2) {
        sc = rsqrtf(bn_var[n] + 1e-5f) * bn_g[n];
        sh = bn_b[n] - bn_mean[n] * sc;
    }
#pragma unroll
    for (int r = 0; r < 8; ++r) {
        int   m   = m0 + wm * 16 + hi * 8 + r;
        float val = acc[r] + bv;
        if (MODE == 2) {
            val = val * sc + sh;                         // inference BatchNorm
            val = val * (1.0f / (1.0f + __expf(-val)));  // SiLU
            int b = m / HWSZ, hw = m - b * HWSZ;
            out[((size_t)b * CC + n) * HWSZ + hw] = val; // NCHW
        } else {
            out[(size_t)m * N + n] = val;
        }
    }
}

// ---------------------------------------------------------------------------
// Deformable bilinear sampling + softmax(mask). One wave per (pixel, group),
// lane = channel within group (GC = 32 = wave32). 128B contiguous gathers.
// ---------------------------------------------------------------------------
__global__ __launch_bounds__(256) void sample_kernel(
    const float* __restrict__ v, const float* __restrict__ om,
    unsigned short* __restrict__ s) {
    const int m    = blockIdx.x;
    const int g    = threadIdx.x >> 5;
    const int lane = threadIdx.x & 31;
    const int b  = m / HWSZ;
    const int hw = m - b * HWSZ;
    const int h  = hw / WW;
    const int w  = hw - h * WW;

    const float* omp = om + (size_t)m * NOM;
    // softmax over P = 9 mask logits (uniform per wave)
    float lgt[9], mx = -1e30f;
#pragma unroll
    for (int p = 0; p < 9; ++p) { lgt[p] = omp[144 + g * 9 + p]; mx = fmaxf(mx, lgt[p]); }
    float sum = 0.f;
#pragma unroll
    for (int p = 0; p < 9; ++p) { lgt[p] = __expf(lgt[p] - mx); sum += lgt[p]; }
    const float inv = 1.0f / sum;

    const float* vb = v + (size_t)b * HWSZ * CC + g * GCH + lane;
    float acc = 0.f;
#pragma unroll
    for (int p = 0; p < 9; ++p) {
        float ox = omp[g * 18 + 2 * p];
        float oy = omp[g * 18 + 2 * p + 1];
        // padded-grid math collapses to unpadded coords (zero ring contributes 0)
        float px = (float)(w + (p / 3) - 1) + ox;
        float py = (float)(h + (p % 3) - 1) + oy;
        float fx = floorf(px), fy = floorf(py);
        float wx = px - fx,   wy = py - fy;
        int x0 = (int)fx, y0 = (int)fy;
        float mw = lgt[p] * inv;
        float w00 = (1.f - wx) * (1.f - wy) * mw;
        float w10 = wx * (1.f - wy) * mw;
        float w01 = (1.f - wx) * wy * mw;
        float w11 = wx * wy * mw;
        if ((unsigned)x0 < WW && (unsigned)y0 < HH)
            acc = fmaf(w00, vb[(size_t)(y0 * WW + x0) * CC], acc);
        if ((unsigned)(x0 + 1) < WW && (unsigned)y0 < HH)
            acc = fmaf(w10, vb[(size_t)(y0 * WW + x0 + 1) * CC], acc);
        if ((unsigned)x0 < WW && (unsigned)(y0 + 1) < HH)
            acc = fmaf(w01, vb[(size_t)((y0 + 1) * WW + x0) * CC], acc);
        if ((unsigned)(x0 + 1) < WW && (unsigned)(y0 + 1) < HH)
            acc = fmaf(w11, vb[(size_t)((y0 + 1) * WW + x0 + 1) * CC], acc);
    }
    s[(size_t)m * CC + g * GCH + lane] = f2bf(acc);
}

// ---------------------------------------------------------------------------
// Host-side orchestration
// ---------------------------------------------------------------------------
extern "C" void kernel_launch(void* const* d_in, const int* in_sizes, int n_in,
                              void* d_out, int out_size, void* d_ws, size_t ws_size,
                              hipStream_t stream) {
    (void)in_sizes; (void)n_in; (void)out_size; (void)ws_size;
    const float* x       = (const float*)d_in[0];
    const float* w_in    = (const float*)d_in[1];
    const float* b_in    = (const float*)d_in[2];
    const float* dw_w    = (const float*)d_in[3];
    const float* dw_b    = (const float*)d_in[4];
    const float* ln_g    = (const float*)d_in[5];
    const float* ln_b    = (const float*)d_in[6];
    const float* w_off   = (const float*)d_in[7];
    const float* b_off   = (const float*)d_in[8];
    const float* w_mask  = (const float*)d_in[9];
    const float* b_mask  = (const float*)d_in[10];
    const float* w_out   = (const float*)d_in[11];
    const float* b_out   = (const float*)d_in[12];
    const float* bn_g    = (const float*)d_in[13];
    const float* bn_b    = (const float*)d_in[14];
    const float* bn_mean = (const float*)d_in[15];
    const float* bn_var  = (const float*)d_in[16];
    float* y = (float*)d_out;

    // Workspace layout (256B aligned)
    char* base = (char*)d_ws;
    size_t off = 0;
    auto take = [&](size_t bytes) {
        void* p = base + off;
        off += (bytes + 255) & ~(size_t)255;
        return p;
    };
    unsigned short* wInT   = (unsigned short*)take((size_t)CC * CC * 2);
    unsigned short* wOutT  = (unsigned short*)take((size_t)CC * CC * 2);
    unsigned short* wOMT   = (unsigned short*)take((size_t)NOM * CC * 2);
    float*          biasOM = (float*)take(NOM * 4);
    unsigned short* xT     = (unsigned short*)take((size_t)MTOT * CC * 2); // reused as sampled
    unsigned short* fbf    = (unsigned short*)take((size_t)MTOT * CC * 2);
    float*          vfeat  = (float*)take((size_t)MTOT * CC * 4);
    float*          ftmp   = (float*)take((size_t)MTOT * CC * 4);          // reused as om
    unsigned short* samp   = xT;     // alias: xT dead after input-proj GEMM
    float*          om     = ftmp;   // alias: ftmp dead after LN kernel

    // 1. Weight prep
    prep_wT_kernel<<<256, 256, 0, stream>>>(w_in, wInT, 256);
    prep_wT_kernel<<<256, 256, 0, stream>>>(w_out, wOutT, 256);
    prep_wT_kernel<<<144, 256, 0, stream>>>(w_off, wOMT, 144);
    prep_wT_kernel<<<72, 256, 0, stream>>>(w_mask, wOMT + 144 * 256, 72);
    prep_om_tail_kernel<<<8, 256, 0, stream>>>(b_off, b_mask, wOMT, biasOM);

    // 2. x -> NHWC bf16
    transpose_x_kernel<<<dim3(HWSZ / 32, CC / 32, BB), 256, 0, stream>>>(x, xT);

    // 3. depthwise conv -> fp32 tmp; LN + GELU -> bf16 f
    dw_conv_kernel<<<dim3(HH, BB * CC), 64, 0, stream>>>(x, dw_w, dw_b, ftmp);
    ln_gelu_kernel<<<MTOT, 256, 0, stream>>>(ftmp, ln_g, ln_b, fbf);

    // 4. input projection GEMM: v = xT @ w_in + b_in   (fp32 out, NHWC)
    gemm_bf16_kernel<0><<<dim3(CC / 32, MTOT / 64), 256, 0, stream>>>(
        xT, wInT, b_in, vfeat, CC, nullptr, nullptr, nullptr, nullptr);

    // 5. fused offset+mask GEMM: om = f @ [w_off|w_mask] + bias  (fp32, [M,224])
    gemm_bf16_kernel<0><<<dim3(NOM / 32, MTOT / 64), 256, 0, stream>>>(
        fbf, wOMT, biasOM, om, NOM, nullptr, nullptr, nullptr, nullptr);

    // 6. deformable sampling -> bf16 sampled features (overwrites xT buffer)
    sample_kernel<<<MTOT, 256, 0, stream>>>(vfeat, om, samp);

    // 7. output projection + BN + SiLU, NCHW scatter into d_out
    gemm_bf16_kernel<2><<<dim3(CC / 32, MTOT / 64), 256, 0, stream>>>(
        samp, wOutT, b_out, y, CC, bn_g, bn_b, bn_mean, bn_var);
}
